// nrecGNN_prop_85418309583443
// MI455X (gfx1250) — matrix operand order repaired and checked by reading, not compiled, not run on hardware
//
#include <hip/hip_runtime.h>
#include <hip/hip_bf16.h>
#include <math.h>

typedef __attribute__((ext_vector_type(16))) _Float16 v16h;
typedef __attribute__((ext_vector_type(8)))  _Float16 v8h;
typedef __attribute__((ext_vector_type(8)))  float    v8f;

#define N_NODES 50000
#define N_HOPS  4
#define N_EDGES 800000
#define NFEAT   512
#define HIDDEN  512
#define NCLASS  128
#define ATT_HID 64

// ---------------------------------------------------------------------------
// Async copy helpers (CDNA5 GLOBAL_LOAD_ASYNC_TO_LDS, tracked by ASYNCcnt)
// ---------------------------------------------------------------------------
__device__ __forceinline__ unsigned lds_off(const void* p) {
    // generic pointer to LDS: low 32 bits are the wave-relative LDS address
    return (unsigned)(uintptr_t)p;
}

__device__ __forceinline__ void async_b128(unsigned lds_addr, const void* g) {
    asm volatile("global_load_async_to_lds_b128 %0, %1, off"
                 :: "v"(lds_addr), "v"(g) : "memory");
}

__device__ __forceinline__ void wait_async0() {
    asm volatile("s_wait_asynccnt 0x0" ::: "memory");
}

// ---------------------------------------------------------------------------
// Generic tiled WMMA GEMM: C[M,N] = act(A[M,K] @ BT[N,K]^T (+ bias))
// A: f16 (async-staged) or f32 (VGPR-convert staged). BT: f16, always async.
// Block tile 128x128, K-step 32, 256 threads = 8 waves, wave tile 32x64.
// Double-buffered LDS pipeline overlaps staging of tile t+1 with WMMA on t.
// ---------------------------------------------------------------------------
template <typename TA, bool OUT_HALF, bool RELU, bool BIAS>
__global__ __launch_bounds__(256) void gemm_wmma(
    const TA* __restrict__ A, const _Float16* __restrict__ BT,
    void* __restrict__ Cout, const float* __restrict__ bias,
    int M, int N, int K)
{
    constexpr bool A_HALF = (sizeof(TA) == 2);
    __shared__ __align__(16) _Float16 As[2][128 * 32];  // [row][k]
    __shared__ __align__(16) _Float16 Bs[2][128 * 32];  // [n][k]

    const int tid  = threadIdx.x;
    const int lane = tid & 31;
    const int wave = tid >> 5;
    const int wm   = wave & 3;   // 0..3, 32 rows each
    const int wn   = wave >> 2;  // 0..1, 64 cols each
    const int m0   = blockIdx.y * 128;
    const int n0   = blockIdx.x * 128;
    const int ml   = lane & 15;
    const int hi   = lane >> 4;

    const v8f vzero = {0.f, 0.f, 0.f, 0.f, 0.f, 0.f, 0.f, 0.f};
    v8f acc[2][4];
#pragma unroll
    for (int mi = 0; mi < 2; ++mi)
#pragma unroll
        for (int ni = 0; ni < 4; ++ni) acc[mi][ni] = vzero;

    // staging assignment: thread t handles row = t>>1, halves [(t&1)*16, +16)
    const int st_r  = tid >> 1;
    const int st_k  = (tid & 1) * 16;
    int gA = m0 + st_r; if (gA > M - 1) gA = M - 1;   // clamp (garbage rows unstored)
    int gB = n0 + st_r; if (gB > N - 1) gB = N - 1;
    const TA*       srcA = A  + (size_t)gA * K + st_k;
    const _Float16* srcB = BT + (size_t)gB * K + st_k;

    auto stage = [&](int t, int buf) {
        const int kk = t << 5;
        if (A_HALF) {
            unsigned d = lds_off(&As[buf][st_r * 32 + st_k]);
            async_b128(d,      (const _Float16*)srcA + kk);
            async_b128(d + 16, (const _Float16*)srcA + kk + 8);
        } else {
            _Float16* dst = &As[buf][st_r * 32 + st_k];
            const TA* s = srcA + kk;
#pragma unroll
            for (int j = 0; j < 16; ++j) dst[j] = (_Float16)(float)s[j];
        }
        unsigned d = lds_off(&Bs[buf][st_r * 32 + st_k]);
        async_b128(d,      srcB + kk);
        async_b128(d + 16, srcB + kk + 8);
    };

    const int nk = K >> 5;
    stage(0, 0);

    for (int t = 0; t < nk; ++t) {
        wait_async0();          // own async copies for tile t have landed
        __syncthreads();        // everyone's copies visible (also waits DScnt)
        if (t + 1 < nk) stage(t + 1, (t + 1) & 1);  // overlap with compute

        const _Float16* Ab = As[t & 1];
        const _Float16* Bb = Bs[t & 1];

        union AF { v16h v; v8h h[2]; };
        AF af[2];
#pragma unroll
        for (int mi = 0; mi < 2; ++mi) {
            const int r = wm * 32 + mi * 16 + ml;
            af[mi].h[0] = *(const v8h*)&Ab[r * 32 + hi * 8];       // K 0-7 / 8-15
            af[mi].h[1] = *(const v8h*)&Ab[r * 32 + 16 + hi * 8];  // K 16-23 / 24-31
        }
#pragma unroll
        for (int ni = 0; ni < 4; ++ni) {
            const int n = wn * 64 + ni * 16 + ml;
            AF bf;
            bf.h[0] = *(const v8h*)&Bb[n * 32 + hi * 8];       // K 0-7  / 8-15
            bf.h[1] = *(const v8h*)&Bb[n * 32 + 16 + hi * 8];  // K 16-23 / 24-31
#pragma unroll
            for (int mi = 0; mi < 2; ++mi)
                acc[mi][ni] = __builtin_amdgcn_wmma_f32_16x16x32_f16(
                    false, af[mi].v, false, bf.v, (short)0, acc[mi][ni],
                    false, false);
        }
    }

    // epilogue (C/D layout: VGPR r -> row hi*8+r, col lane%16)
#pragma unroll
    for (int ni = 0; ni < 4; ++ni) {
        const int col = n0 + wn * 64 + ni * 16 + ml;
        if (col >= N) continue;
        const float bv = BIAS ? bias[col] : 0.f;
#pragma unroll
        for (int mi = 0; mi < 2; ++mi) {
#pragma unroll
            for (int r = 0; r < 8; ++r) {
                const int row = m0 + wm * 32 + mi * 16 + hi * 8 + r;
                if (row < M) {
                    float v = acc[mi][ni][r] + bv;
                    if (RELU) v = v > 0.f ? v : 0.f;
                    if (OUT_HALF)
                        ((_Float16*)Cout)[(size_t)row * N + col] = (_Float16)v;
                    else
                        ((float*)Cout)[(size_t)row * N + col] = v;
                }
            }
        }
    }
}

// ---------------------------------------------------------------------------
// helpers
// ---------------------------------------------------------------------------
__global__ void cvt_f16_kernel(const float* __restrict__ in,
                               _Float16* __restrict__ out, int n) {
    int i = blockIdx.x * 256 + threadIdx.x;
    if (i < n) out[i] = (_Float16)in[i];
}

// B[K,N] f32 -> BT[N,K] f16
__global__ void transpose_cvt_kernel(const float* __restrict__ B,
                                     _Float16* __restrict__ BT, int K, int N) {
    int i = blockIdx.x * 256 + threadIdx.x;
    if (i < K * N) {
        int k = i / N, n = i - k * N;
        BT[(size_t)n * K + k] = (_Float16)B[i];
    }
}

__global__ void fill0_kernel(float* __restrict__ p, int n) {
    int i = blockIdx.x * 256 + threadIdx.x;
    if (i < n) p[i] = 0.f;
}

__global__ void gather_rows_kernel(const float* __restrict__ z,
                                   const int* __restrict__ idx,
                                   float* __restrict__ anchor, int total) {
    int i = blockIdx.x * 256 + threadIdx.x;
    if (i < total) {
        int r = i >> 7, c = i & 127;
        anchor[i] = z[(size_t)idx[r] * NCLASS + c];
    }
}

// one wave per edge: agg[row] += val * z[col]  (128 floats, 4 per lane)
__global__ __launch_bounds__(256) void spmm_kernel(
    const float* __restrict__ vals, const int* __restrict__ rows,
    const int* __restrict__ cols, const float* __restrict__ z,
    float* __restrict__ agg, int nedges)
{
    const int e = blockIdx.x * 8 + (threadIdx.x >> 5);
    if (e >= nedges) return;
    const int lane = threadIdx.x & 31;
    const int r = rows[e];
    const int c = cols[e];
    const float v = vals[e];
    const float4 zz = *(const float4*)(z + (size_t)c * NCLASS + lane * 4);
    float* dst = agg + (size_t)r * NCLASS + lane * 4;
    atomicAdd(dst + 0, v * zz.x);
    atomicAdd(dst + 1, v * zz.y);
    atomicAdd(dst + 2, v * zz.z);
    atomicAdd(dst + 3, v * zz.w);
}

// scores[row,h] = va . tanh(qproj[row] + sproj[row] + ba)   (one wave / row)
__global__ __launch_bounds__(256) void scores_kernel(
    const float* __restrict__ qproj, const float* __restrict__ sproj,
    const float* __restrict__ ba, const float* __restrict__ va,
    float* __restrict__ scores, int nrows, int h)
{
    const int row = blockIdx.x * 8 + (threadIdx.x >> 5);
    if (row >= nrows) return;
    const int lane = threadIdx.x & 31;
    float s = 0.f;
#pragma unroll
    for (int j = lane; j < ATT_HID; j += 32) {
        const float t = tanhf(qproj[(size_t)row * ATT_HID + j] +
                              sproj[(size_t)row * ATT_HID + j] + ba[j]);
        s += t * va[j];
    }
#pragma unroll
    for (int off = 16; off > 0; off >>= 1) s += __shfl_xor(s, off, 32);
    if (lane == 0) scores[row * 5 + h] = s;
}

// softmax(scores) -> weighted hop combine -> log_softmax (one wave / row)
__global__ __launch_bounds__(256) void finalize_kernel(
    const float* __restrict__ anchor, const float* __restrict__ emb,
    const float* __restrict__ scores, float* __restrict__ out, int nrows)
{
    const int row = blockIdx.x * 8 + (threadIdx.x >> 5);
    if (row >= nrows) return;
    const int lane = threadIdx.x & 31;

    float sc[5];
    float mx = -1e30f;
#pragma unroll
    for (int h = 0; h < 5; ++h) { sc[h] = scores[row * 5 + h]; mx = fmaxf(mx, sc[h]); }
    float se = 0.f;
#pragma unroll
    for (int h = 0; h < 5; ++h) { sc[h] = expf(sc[h] - mx); se += sc[h]; }
    const float inv = 1.f / se;

    const size_t rb = (size_t)row * NCLASS + lane * 4;
    const float4 a = *(const float4*)(anchor + rb);
    float o[4] = {sc[0] * inv * a.x, sc[0] * inv * a.y,
                  sc[0] * inv * a.z, sc[0] * inv * a.w};
#pragma unroll
    for (int h = 0; h < N_HOPS; ++h) {
        const float4 e = *(const float4*)(emb + (size_t)h * nrows * NCLASS + rb);
        const float w = sc[h + 1] * inv;
        o[0] += w * e.x; o[1] += w * e.y; o[2] += w * e.z; o[3] += w * e.w;
    }
    float m = fmaxf(fmaxf(o[0], o[1]), fmaxf(o[2], o[3]));
#pragma unroll
    for (int off = 16; off > 0; off >>= 1) m = fmaxf(m, __shfl_xor(m, off, 32));
    float s = expf(o[0] - m) + expf(o[1] - m) + expf(o[2] - m) + expf(o[3] - m);
#pragma unroll
    for (int off = 16; off > 0; off >>= 1) s += __shfl_xor(s, off, 32);
    const float lse = m + logf(s);
    float4 r4 = {o[0] - lse, o[1] - lse, o[2] - lse, o[3] - lse};
    *(float4*)(out + rb) = r4;
}

// ---------------------------------------------------------------------------
// launch
// ---------------------------------------------------------------------------
extern "C" void kernel_launch(void* const* d_in, const int* in_sizes, int n_in,
                              void* d_out, int out_size, void* d_ws, size_t ws_size,
                              hipStream_t stream) {
    const float* x   = (const float*)d_in[0];
    const float* W1  = (const float*)d_in[1];
    const float* W2  = (const float*)d_in[2];
    const float* b2  = (const float*)d_in[3];
    const float* Wg  = (const float*)d_in[4];
    const float* Wa  = (const float*)d_in[5];   // [256,64]
    const float* ba  = (const float*)d_in[6];
    const float* va  = (const float*)d_in[7];
    const float* adj_vals = (const float*)d_in[8];
    const int*   adj_idx  = (const int*)d_in[9];
    const int*   idx      = (const int*)d_in[10];

    // workspace layout
    char* ws = (char*)d_ws;
    _Float16* xh   = (_Float16*)ws;  ws += (size_t)N_NODES * NFEAT * 2;         // 51.2 MB
    _Float16* h1   = (_Float16*)ws;  ws += (size_t)N_NODES * HIDDEN * 2;        // 51.2 MB
    _Float16* W1T  = (_Float16*)ws;  ws += (size_t)HIDDEN * NFEAT * 2;          // 0.5 MB
    _Float16* W2T  = (_Float16*)ws;  ws += (size_t)NCLASS * HIDDEN * 2;         // 128 KB
    _Float16* WgT  = (_Float16*)ws;  ws += (size_t)NCLASS * NCLASS * 2;         // 32 KB
    _Float16* WaTT = (_Float16*)ws;  ws += (size_t)ATT_HID * NCLASS * 2;        // 16 KB
    _Float16* WaBT = (_Float16*)ws;  ws += (size_t)ATT_HID * NCLASS * 2;        // 16 KB
    float* z      = (float*)ws;      ws += (size_t)N_NODES * NCLASS * 4;        // 25.6 MB
    float* anchor = (float*)ws;      ws += (size_t)N_NODES * NCLASS * 4;        // 25.6 MB
    float* agg    = (float*)ws;      ws += (size_t)N_NODES * NCLASS * 4;        // 25.6 MB
    float* emb    = (float*)ws;      ws += (size_t)N_HOPS * N_NODES * NCLASS * 4; // 102.4 MB
    float* qproj  = (float*)ws;      ws += (size_t)N_NODES * ATT_HID * 4;       // 12.8 MB
    float* sproj  = (float*)ws;      ws += (size_t)N_NODES * ATT_HID * 4;       // 12.8 MB
    float* scores = (float*)ws;      ws += (size_t)N_NODES * 5 * 4;             // 1 MB

    const dim3 blk(256);
    const int mblk = (N_NODES + 127) / 128;  // 391

    // 0) precision/layout prep: x -> f16, weights -> f16 transposed [N,K]
    cvt_f16_kernel<<<(N_NODES * NFEAT + 255) / 256, blk, 0, stream>>>(
        x, xh, N_NODES * NFEAT);
    transpose_cvt_kernel<<<(NFEAT * HIDDEN + 255) / 256, blk, 0, stream>>>(
        W1, W1T, NFEAT, HIDDEN);
    transpose_cvt_kernel<<<(HIDDEN * NCLASS + 255) / 256, blk, 0, stream>>>(
        W2, W2T, HIDDEN, NCLASS);
    transpose_cvt_kernel<<<(NCLASS * NCLASS + 255) / 256, blk, 0, stream>>>(
        Wg, WgT, NCLASS, NCLASS);
    transpose_cvt_kernel<<<(NCLASS * ATT_HID + 255) / 256, blk, 0, stream>>>(
        Wa, WaTT, NCLASS, ATT_HID);
    transpose_cvt_kernel<<<(NCLASS * ATT_HID + 255) / 256, blk, 0, stream>>>(
        Wa + (size_t)NCLASS * ATT_HID, WaBT, NCLASS, ATT_HID);

    // 1) h1 = relu(x @ W1)   [50000,512] f16  (fully async-staged)
    gemm_wmma<_Float16, true, true, false><<<dim3(HIDDEN / 128, mblk), blk, 0, stream>>>(
        xh, W1T, h1, nullptr, N_NODES, HIDDEN, NFEAT);

    // 2) z = h1 @ W2 + b2    [50000,128] f32  (fully async-staged)
    gemm_wmma<_Float16, false, false, true><<<dim3(1, mblk), blk, 0, stream>>>(
        h1, W2T, z, b2, N_NODES, NCLASS, HIDDEN);

    // 3) anchor = z[idx]
    gather_rows_kernel<<<(N_NODES * NCLASS + 255) / 256, blk, 0, stream>>>(
        z, idx, anchor, N_NODES * NCLASS);

    // 4) qproj = anchor @ Wa_top ; sproj = anchor @ Wa_bot ; scores[:,0]
    gemm_wmma<float, false, false, false><<<dim3(1, mblk), blk, 0, stream>>>(
        anchor, WaTT, qproj, nullptr, N_NODES, ATT_HID, NCLASS);
    gemm_wmma<float, false, false, false><<<dim3(1, mblk), blk, 0, stream>>>(
        anchor, WaBT, sproj, nullptr, N_NODES, ATT_HID, NCLASS);
    scores_kernel<<<(N_NODES + 7) / 8, blk, 0, stream>>>(
        qproj, sproj, ba, va, scores, N_NODES, 0);

    // 5) hop loop
    for (int h = 0; h < N_HOPS; ++h) {
        const float* vals = adj_vals + (size_t)h * N_EDGES;
        const int*   rows = adj_idx + (size_t)h * 2 * N_EDGES;
        const int*   cols = rows + N_EDGES;
        float* emb_h = emb + (size_t)h * N_NODES * NCLASS;

        fill0_kernel<<<(N_NODES * NCLASS + 255) / 256, blk, 0, stream>>>(
            agg, N_NODES * NCLASS);
        spmm_kernel<<<(N_EDGES + 7) / 8, blk, 0, stream>>>(
            vals, rows, cols, z, agg, N_EDGES);
        gemm_wmma<float, false, false, false><<<dim3(1, mblk), blk, 0, stream>>>(
            agg, WgT, emb_h, nullptr, N_NODES, NCLASS, NCLASS);
        gemm_wmma<float, false, false, false><<<dim3(1, mblk), blk, 0, stream>>>(
            emb_h, WaBT, sproj, nullptr, N_NODES, ATT_HID, NCLASS);
        scores_kernel<<<(N_NODES + 7) / 8, blk, 0, stream>>>(
            qproj, sproj, ba, va, scores, N_NODES, h + 1);
    }

    // 6) attention combine + log_softmax -> out [50000,128]
    finalize_kernel<<<(N_NODES + 7) / 8, blk, 0, stream>>>(
        anchor, emb, scores, (float*)d_out, N_NODES);
}
